// GAT_EED_Transformer_89240830476325
// MI455X (gfx1250) — matrix operand order repaired and checked by reading, not compiled, not run on hardware
//
#include <hip/hip_runtime.h>

// ---------------------------------------------------------------------------
// CDNA5 (gfx1250) GAT + seq1-Transformer pipeline, bf16 WMMA GEMMs.
// Round 3: FF kernel widened to 128-row blocks (8 waves) + weight prefetch
// (global_prefetch_b8). Fragment loads remain vectorized ds_load_b128 with
// padded LDS; weights kept in [out,in] layout so B fragments are k-contiguous.
// ---------------------------------------------------------------------------

#define DEV __device__ __forceinline__

typedef __attribute__((ext_vector_type(16))) __bf16 v16bf;
typedef __attribute__((ext_vector_type(8)))  float  v8f;

// ---- bf16 helpers (manual RNE, no header deps) ----------------------------
DEV __bf16 f2bf(float f) {
  unsigned u = __float_as_uint(f);
  unsigned r = u + 0x7FFFu + ((u >> 16) & 1u);
  unsigned short s = (unsigned short)(r >> 16);
  __bf16 b; __builtin_memcpy(&b, &s, 2); return b;
}

union FragCvt { uint4 q[2]; v16bf v; };

// ---- WMMA fragment loaders from LDS (CDNA5 ISA 7.12.2 layouts) ------------
// A 16x32 bf16: lane holds row m=lane&15; element e: K = e' pattern =
//   {0..7}+8*hi (e<8), {16..23}+8*hi (e>=8)  -> two contiguous 16B runs.
DEV v16bf frag_a_lds(const __bf16* s, int ld, int row0, int k0, int lane) {
  int m = row0 + (lane & 15);
  int khi8 = ((lane >> 4) & 1) << 3;
  const __bf16* p = s + m * ld + k0 + khi8;
  FragCvt u;
  u.q[0] = *reinterpret_cast<const uint4*>(p);       // K = k0+khi8 + 0..7
  u.q[1] = *reinterpret_cast<const uint4*>(p + 16);  // K = k0+khi8 + 16..23
  return u.v;
}
// B 32x16 bf16 stored k-contiguous per n: sB[n*ld + k]; element e:
//   K = {0..7}+8*hi (e<8), {16..23}+8*hi (e>=8)  -> two contiguous 16B runs.
DEV v16bf frag_b_lds(const __bf16* s, int ld, int k0, int col0, int lane) {
  int n = col0 + (lane & 15);
  int khi16 = ((lane >> 4) & 1) << 3;
  const __bf16* p = s + n * ld + k0 + khi16;
  FragCvt u;
  u.q[0] = *reinterpret_cast<const uint4*>(p);
  u.q[1] = *reinterpret_cast<const uint4*>(p + 16);
  return u.v;
}

// ---- orderable-uint float max mapping -------------------------------------
DEV unsigned fmapu(float f) {
  unsigned u = __float_as_uint(f);
  return (u & 0x80000000u) ? ~u : (u | 0x80000000u);
}
DEV float funmap(unsigned m) {
  return (m & 0x80000000u) ? __uint_as_float(m & 0x7FFFFFFFu)
                           : __uint_as_float(~m);
}

// ===========================================================================
// Small utility kernels
// ===========================================================================
__global__ void fill_u32_kernel(unsigned* p, unsigned v, int n) {
  int i = blockIdx.x * blockDim.x + threadIdx.x;
  if (i < n) p[i] = v;
}
__global__ void cvt_f32_bf16_kernel(const float* in, __bf16* out, int n) {
  int i = blockIdx.x * blockDim.x + threadIdx.x;
  if (i < n) out[i] = f2bf(in[i]);
}

// ===========================================================================
// Generic tiled WMMA GEMM:  C[M,Nc] = A[M,K] @ W^T (+bias)
// W given bf16 in original [Nc, K] (out,in) layout.
// block = 128 threads (4 waves), block tile 64 x BN, K step 32.
// LDS padded: ld = 40 halfs (80B rows -> conflict-free b128 frags).
// ===========================================================================
template <int BN>
__launch_bounds__(128)
__global__ void gemm_bf16_kernel(const __bf16* __restrict__ A,
                                 const __bf16* __restrict__ W,
                                 const float* __restrict__ bias,
                                 float* __restrict__ Cf,
                                 __bf16* __restrict__ Cbf,
                                 int M, int K, int Nc) {
  constexpr int NT = BN / 16;
  constexpr int LDP = 40;
  __shared__ __attribute__((aligned(16))) __bf16 sA[64 * LDP];
  __shared__ __attribute__((aligned(16))) __bf16 sB[BN * LDP];
  const int tid = threadIdx.x, wave = tid >> 5, lane = tid & 31;
  const int khi = (lane >> 4) & 1;
  const int rowblk = blockIdx.x * 64, colblk = blockIdx.y * BN;
  const uint4* Ag = reinterpret_cast<const uint4*>(A);
  const uint4* Wg = reinterpret_cast<const uint4*>(W);

  v8f acc[NT];
#pragma unroll
  for (int t = 0; t < NT; ++t)
#pragma unroll
    for (int j = 0; j < 8; ++j) acc[t][j] = 0.0f;

  for (int k0 = 0; k0 < K; k0 += 32) {
    __syncthreads();
    // stage A tile: 64 rows x 32 k  (4 uint4-chunks of 8 bf16 per row)
    for (int i = tid; i < 64 * 4; i += 128) {
      int r = i >> 2, c = i & 3, gr = rowblk + r;
      uint4 v = {0u, 0u, 0u, 0u};
      if (gr < M) v = Ag[((gr * K + k0) >> 3) + c];
      *reinterpret_cast<uint4*>(&sA[r * LDP + c * 8]) = v;
    }
    // stage B tile (k-contiguous): BN rows (n) x 32 k
    for (int i = tid; i < BN * 4; i += 128) {
      int r = i >> 2, c = i & 3;
      uint4 v = Wg[(((colblk + r) * K + k0) >> 3) + c];
      *reinterpret_cast<uint4*>(&sB[r * LDP + c * 8]) = v;
    }
    __syncthreads();
    v16bf af = frag_a_lds(sA, LDP, 16 * wave, 0, lane);
#pragma unroll
    for (int t = 0; t < NT; ++t) {
      v16bf bf_ = frag_b_lds(sB, LDP, 0, 16 * t, lane);
      acc[t] = __builtin_amdgcn_wmma_f32_16x16x32_bf16(
          false, af, false, bf_, (short)0, acc[t], false, false);
    }
  }
#pragma unroll
  for (int t = 0; t < NT; ++t) {
    int n = colblk + 16 * t + (lane & 15);
    float bb = bias ? bias[n] : 0.0f;
#pragma unroll
    for (int j = 0; j < 8; ++j) {
      int m = 16 * wave + j + 8 * khi;
      int gr = rowblk + m;
      if (gr < M) {
        float v = acc[t][j] + bb;
        if (Cf)  Cf[gr * Nc + n] = v;
        if (Cbf) Cbf[gr * Nc + n] = f2bf(v);
      }
    }
  }
}

// ===========================================================================
// Fused FFN: Y = relu(A @ W1^T + b1) @ W2^T + b2   (d=64, FF=2048)
// W1 bf16 [2048,64], W2 bf16 [64,2048] (original layouts).
// 128-row blocks (8 waves / 256 threads); 32 chunks of 64 FF units kept
// entirely in LDS; next chunk's weights prefetched (global_prefetch_b8).
// LDS ld = 72 halfs (144B rows -> conflict-free b128 fragments).
// ===========================================================================
__launch_bounds__(256)
__global__ void ff_fused_kernel(const __bf16* __restrict__ Abf,   // [M,64]
                                const __bf16* __restrict__ W1,    // [2048,64]
                                const float* __restrict__ bias1,  // [2048]
                                const __bf16* __restrict__ W2,    // [64,2048]
                                const float* __restrict__ bias2,  // [64]
                                float* __restrict__ Y,            // [M,64]
                                int M) {
  constexpr int LDP = 72;
  __shared__ __attribute__((aligned(16))) __bf16 sA[128 * LDP];
  __shared__ __attribute__((aligned(16))) __bf16 sW1[64 * LDP];   // [n_ff][k_d]
  __shared__ __attribute__((aligned(16))) __bf16 sMid[128 * LDP]; // [m][k_ff]
  __shared__ __attribute__((aligned(16))) __bf16 sW2[64 * LDP];   // [n_out][k_ff]
  const int tid = threadIdx.x, wave = tid >> 5, lane = tid & 31;
  const int khi = (lane >> 4) & 1;
  const int rowblk = blockIdx.x * 128;
  const uint4* Agq = reinterpret_cast<const uint4*>(Abf);
  const uint4* W1q = reinterpret_cast<const uint4*>(W1);
  const uint4* W2q = reinterpret_cast<const uint4*>(W2);

  // stage A: 128 rows x 64 cols = 8 uint4-chunks per row
  for (int i = tid; i < 128 * 8; i += 256) {
    int r = i >> 3, c = i & 7, gr = rowblk + r;
    uint4 v = {0u, 0u, 0u, 0u};
    if (gr < M) v = Agq[gr * 8 + c];
    *reinterpret_cast<uint4*>(&sA[r * LDP + c * 8]) = v;
  }

  v8f acc[4];
#pragma unroll
  for (int t = 0; t < 4; ++t)
#pragma unroll
    for (int j = 0; j < 8; ++j) acc[t][j] = 0.0f;

  for (int ch = 0; ch < 32; ++ch) {
    const int fb = ch * 64;
    __syncthreads();  // prior GEMM2 reads of sMid/sW2 done (covers sA store too)
    for (int i = tid; i < 64 * 8; i += 256) {
      int r = i >> 3, c = i & 7;
      // W1 rows are ff units: sW1[n_ff][k_d]
      *reinterpret_cast<uint4*>(&sW1[r * LDP + c * 8]) = W1q[(fb + r) * 8 + c];
      // W2 rows are outputs, cols are ff: sW2[n_out][k_ff_local]
      *reinterpret_cast<uint4*>(&sW2[r * LDP + c * 8]) = W2q[((r * 2048 + fb) >> 3) + c];
    }
    // prefetch next chunk's weights while this chunk computes
    if (ch < 31) {
      int fb2 = fb + 64;
      if (tid < 64) {
        __builtin_prefetch(&W1[(fb2 + tid) * 64], 0, 0);
      } else if (tid < 128) {
        int r = tid - 64;
        __builtin_prefetch(&W2[r * 2048 + fb2], 0, 0);
      }
    }
    __syncthreads();

    v8f mid[4];
#pragma unroll
    for (int t = 0; t < 4; ++t)
#pragma unroll
      for (int j = 0; j < 8; ++j) mid[t][j] = 0.0f;

#pragma unroll
    for (int ks = 0; ks < 2; ++ks) {
      v16bf a = frag_a_lds(sA, LDP, 16 * wave, 32 * ks, lane);
#pragma unroll
      for (int t = 0; t < 4; ++t) {
        v16bf b = frag_b_lds(sW1, LDP, 32 * ks, 16 * t, lane);
        mid[t] = __builtin_amdgcn_wmma_f32_16x16x32_bf16(
            false, a, false, b, (short)0, mid[t], false, false);
      }
    }
    // bias1 + relu, stage bf16 intermediate tile (row-major [m][k_ff])
#pragma unroll
    for (int t = 0; t < 4; ++t) {
      int n = 16 * t + (lane & 15);
      float bb = bias1[fb + n];
#pragma unroll
      for (int j = 0; j < 8; ++j) {
        int m = 16 * wave + j + 8 * khi;
        float v = mid[t][j] + bb;
        v = v > 0.0f ? v : 0.0f;
        sMid[m * LDP + n] = f2bf(v);
      }
    }
    __syncthreads();
#pragma unroll
    for (int ks = 0; ks < 2; ++ks) {
      v16bf a2 = frag_a_lds(sMid, LDP, 16 * wave, 32 * ks, lane);
#pragma unroll
      for (int t = 0; t < 4; ++t) {
        v16bf b2f = frag_b_lds(sW2, LDP, 32 * ks, 16 * t, lane);
        acc[t] = __builtin_amdgcn_wmma_f32_16x16x32_bf16(
            false, a2, false, b2f, (short)0, acc[t], false, false);
      }
    }
  }
#pragma unroll
  for (int t = 0; t < 4; ++t) {
    int n = 16 * t + (lane & 15);
    float bb = bias2[n];
#pragma unroll
    for (int j = 0; j < 8; ++j) {
      int m = 16 * wave + j + 8 * khi;
      int gr = rowblk + m;
      if (gr < M) Y[gr * 64 + n] = acc[t][j] + bb;
    }
  }
}

// ===========================================================================
// GAT kernels
// ===========================================================================
__global__ void gat_scores_kernel(const float* __restrict__ pre,
                                  const float* __restrict__ a_s,
                                  const float* __restrict__ a_d,
                                  float* __restrict__ as_, float* __restrict__ ad_,
                                  int C, int Nn) {
  int i = blockIdx.x * blockDim.x + threadIdx.x;
  if (i >= Nn * 4) return;
  int n = i >> 2, h = i & 3;
  const float* p = pre + (size_t)n * 4 * C + h * C;
  const float* vs = a_s + h * C;
  const float* vd = a_d + h * C;
  float ss = 0.0f, sd = 0.0f;
  for (int c = 0; c < C; ++c) { float x = p[c]; ss += x * vs[c]; sd += x * vd[c]; }
  as_[i] = ss; ad_[i] = sd;
}

DEV void edge_pair(const int* ei, int e, int Eo, int& s, int& d) {
  if (e < Eo) { s = ei[e]; d = ei[Eo + e]; }
  else { s = d = e - Eo; }
}

__global__ void edge_max_kernel(const int* __restrict__ ei,
                                const float* __restrict__ as_,
                                const float* __restrict__ ad_,
                                unsigned* __restrict__ mmax, int Etot, int Eo) {
  int t = blockIdx.x * blockDim.x + threadIdx.x;
  if (t >= Etot * 4) return;
  int e = t >> 2, h = t & 3, s, d;
  edge_pair(ei, e, Eo, s, d);
  float v = as_[s * 4 + h] + ad_[d * 4 + h];
  v = v > 0.0f ? v : 0.2f * v;
  atomicMax(&mmax[d * 4 + h], fmapu(v));
}

__global__ void edge_accum_kernel(const int* __restrict__ ei,
                                  const float* __restrict__ as_,
                                  const float* __restrict__ ad_,
                                  const unsigned* __restrict__ mmax,
                                  const float* __restrict__ pre,
                                  float* __restrict__ denom,
                                  float* __restrict__ acc,
                                  int Etot, int Eo, int C) {
  int t = blockIdx.x * blockDim.x + threadIdx.x;
  if (t >= Etot * 4) return;
  int e = t >> 2, h = t & 3, s, d;
  edge_pair(ei, e, Eo, s, d);
  float v = as_[s * 4 + h] + ad_[d * 4 + h];
  v = v > 0.0f ? v : 0.2f * v;
  float m = funmap(mmax[d * 4 + h]);
  float w = __expf(v - m);
  atomicAdd(&denom[d * 4 + h], w);
  const float* sp = pre + (size_t)s * 4 * C + h * C;
  float* dp = acc + (size_t)d * 4 * C + h * C;
  for (int c = 0; c < C; ++c) atomicAdd(&dp[c], w * sp[c]);
}

__global__ void gat_finalize_kernel(const float* __restrict__ acc,
                                    const float* __restrict__ denom,
                                    const float* __restrict__ bias,
                                    float* __restrict__ Hf, __bf16* __restrict__ Hbf,
                                    int Nn, int Dm, int C) {
  int i = blockIdx.x * blockDim.x + threadIdx.x;
  if (i >= Nn * Dm) return;
  int n = i / Dm, d = i % Dm, h = d / C;
  float v = acc[i] / (denom[n * 4 + h] + 1e-16f) + bias[d];
  v = v > 0.0f ? v : 0.01f * v;   // F.leaky_relu default slope
  Hf[i] = v;
  Hbf[i] = f2bf(v);
}

// ===========================================================================
// Fused residual + LayerNorm (d=64): H = LN(a + y)*g + b ; also bf16 copy
// ===========================================================================
__global__ void ln_residual_kernel(const float* __restrict__ a,
                                   const float* __restrict__ y,
                                   const float* __restrict__ g,
                                   const float* __restrict__ b,
                                   float* __restrict__ Hf, __bf16* __restrict__ Hbf,
                                   int Nn) {
  int n = blockIdx.x * blockDim.x + threadIdx.x;
  if (n >= Nn) return;
  float v[64];
  float mu = 0.0f;
#pragma unroll
  for (int i = 0; i < 64; ++i) {
    float t = a[n * 64 + i] + y[n * 64 + i];
    v[i] = t; mu += t;
  }
  mu *= (1.0f / 64.0f);
  float var = 0.0f;
#pragma unroll
  for (int i = 0; i < 64; ++i) { float d = v[i] - mu; var += d * d; }
  var *= (1.0f / 64.0f);
  float r = rsqrtf(var + 1e-5f);
#pragma unroll
  for (int i = 0; i < 64; ++i) {
    float o = (v[i] - mu) * r * g[i] + b[i];
    Hf[n * 64 + i] = o;
    Hbf[n * 64 + i] = f2bf(o);
  }
}

// ===========================================================================
// Pooling + final FC
// ===========================================================================
__global__ void pool_accum_kernel(const float* __restrict__ H,
                                  const int* __restrict__ batch,
                                  float* __restrict__ pooled, float* __restrict__ cnt,
                                  int Nn) {
  int n = blockIdx.x * blockDim.x + threadIdx.x;
  if (n >= Nn) return;
  int g = batch[n];
  atomicAdd(&cnt[g], 1.0f);
  for (int c = 0; c < 64; ++c) atomicAdd(&pooled[g * 64 + c], H[n * 64 + c]);
}

__global__ void final_fc_kernel(const float* __restrict__ pooled,
                                const float* __restrict__ cnt,
                                const float* __restrict__ fw,
                                const float* __restrict__ fb,
                                float* __restrict__ out) {
  int i = blockIdx.x * blockDim.x + threadIdx.x;
  if (i >= 64 * 16) return;
  int g = i / 16, o = i % 16;
  float cc = cnt[g]; cc = cc > 1.0f ? cc : 1.0f;
  float s = 0.0f;
  for (int d = 0; d < 64; ++d) s += (pooled[g * 64 + d] / cc) * fw[o * 64 + d];
  out[i] = s + fb[o];
}

// ===========================================================================
// Host orchestration
// ===========================================================================
static inline int cdiv(int a, int b) { return (a + b - 1) / b; }

extern "C" void kernel_launch(void* const* d_in, const int* in_sizes, int n_in,
                              void* d_out, int out_size, void* d_ws, size_t ws_size,
                              hipStream_t stream) {
  (void)n_in; (void)out_size; (void)ws_size;
  const int Nn = in_sizes[2];          // 50000 nodes
  const int Eo = in_sizes[1] / 2;      // 1.6M edges
  const int Et = Eo + Nn;              // + self loops
  const int Fin = in_sizes[0] / Nn;    // 128

  // ---- input index map (setup_inputs dict insertion order) ----
  auto F = [&](int i) { return (const float*)d_in[i]; };
  const int* ei    = (const int*)d_in[1];
  const int* batch = (const int*)d_in[2];
  const float* x_f = F(0);
  const int I_FCIN_W = 3, I_FCIN_B = 4;
  const int I_G1W = 5, I_G1AS = 6, I_G1AD = 7, I_G1B = 8;
  const int I_G2W = 9, I_G2AS = 10, I_G2AD = 11, I_G2B = 12;
  int Lbase[4]; Lbase[0] = 13; Lbase[1] = 25; Lbase[2] = 37; Lbase[3] = 55;
  const int I_FCW = 73, I_FCB = 74;
  // per-layer: wv,bv,wo,bo,w1,b1,w2,b2,ln1g,ln1b,ln2g,ln2b[,wv2,bv2,wo2,bo2,ln3g,ln3b]

  // ---- workspace bump allocator ----
  char* wp = (char*)d_ws;
  auto alloc = [&](size_t bytes) -> void* {
    void* p = (void*)wp;
    wp += (bytes + 255) & ~(size_t)255;
    return p;
  };
  __bf16* xbf    = (__bf16*)alloc((size_t)Nn * Fin * 2);
  __bf16* actbf  = (__bf16*)alloc((size_t)Nn * 128 * 2);
  __bf16* tmpbf  = (__bf16*)alloc((size_t)Nn * 64 * 2);
  __bf16* membf  = (__bf16*)alloc((size_t)Nn * 64 * 2);
  float*  PRE    = (float*)alloc((size_t)Nn * 128 * 4);
  float*  ACC    = (float*)alloc((size_t)Nn * 128 * 4);
  float*  Hf     = (float*)alloc((size_t)Nn * 128 * 4);
  float*  Yf     = (float*)alloc((size_t)Nn * 64 * 4);
  float*  MEMf   = (float*)alloc((size_t)Nn * 64 * 4);
  float*  as_    = (float*)alloc((size_t)Nn * 4 * 4);
  float*  ad_    = (float*)alloc((size_t)Nn * 4 * 4);
  float*  denom  = (float*)alloc((size_t)Nn * 4 * 4);
  unsigned* mmax = (unsigned*)alloc((size_t)Nn * 4 * 4);
  float*  pooled = (float*)alloc(64 * 64 * 4);
  float*  cntb   = (float*)alloc(64 * 4);
  // bf16 weights, kept in original [out,in] layout
  __bf16* fcinW  = (__bf16*)alloc(32 * 128 * 2);
  __bf16* g1W    = (__bf16*)alloc(128 * 32 * 2);
  __bf16* g2W    = (__bf16*)alloc(64 * 128 * 2);
  __bf16 *wvW[4], *woW[4], *w1W[4], *w2W[4], *wv2W[4], *wo2W[4];
  for (int l = 0; l < 4; ++l) {
    wvW[l] = (__bf16*)alloc(64 * 64 * 2);
    woW[l] = (__bf16*)alloc(64 * 64 * 2);
    w1W[l] = (__bf16*)alloc(2048 * 64 * 2);
    w2W[l] = (__bf16*)alloc(64 * 2048 * 2);
    if (l >= 2) { wv2W[l] = (__bf16*)alloc(64 * 64 * 2); wo2W[l] = (__bf16*)alloc(64 * 64 * 2); }
    else { wv2W[l] = nullptr; wo2W[l] = nullptr; }
  }

  // ---- launch helpers ----
  auto cvt = [&](const float* w, __bf16* o, int n) {
    cvt_f32_bf16_kernel<<<cdiv(n, 256), 256, 0, stream>>>(w, o, n);
  };
  auto fillz = [&](void* p, int n32) {
    fill_u32_kernel<<<cdiv(n32, 256), 256, 0, stream>>>((unsigned*)p, 0u, n32);
  };
  auto gemm64 = [&](const __bf16* A, const __bf16* W, const float* bias,
                    float* Cf, __bf16* Cbf, int M, int K, int Nc) {
    dim3 g(cdiv(M, 64), Nc / 64);
    gemm_bf16_kernel<64><<<g, 128, 0, stream>>>(A, W, bias, Cf, Cbf, M, K, Nc);
  };

  // ---- 1) weight + input conversion (f32 -> bf16, layout preserved) ----
  cvt(F(I_FCIN_W), fcinW, 32 * 128);
  cvt(F(I_G1W), g1W, 128 * 32);
  cvt(F(I_G2W), g2W, 64 * 128);
  for (int l = 0; l < 4; ++l) {
    int b = Lbase[l];
    cvt(F(b + 0), wvW[l], 64 * 64);
    cvt(F(b + 2), woW[l], 64 * 64);
    cvt(F(b + 4), w1W[l], 2048 * 64);
    cvt(F(b + 6), w2W[l], 64 * 2048);
    if (l >= 2) { cvt(F(b + 12), wv2W[l], 64 * 64); cvt(F(b + 14), wo2W[l], 64 * 64); }
  }
  cvt(x_f, xbf, Nn * Fin);

  // ---- 2) fc_in : h0 = x @ Wfc^T + b  -> bf16 [N,32] ----
  {
    dim3 g(cdiv(Nn, 64), 1);
    gemm_bf16_kernel<32><<<g, 128, 0, stream>>>(xbf, fcinW, F(I_FCIN_B),
                                                nullptr, actbf, Nn, Fin, 32);
  }

  // ---- 3) GAT layers ----
  auto gat_layer = [&](const __bf16* Abf, const __bf16* W, int K, int C,
                       const float* a_s, const float* a_d, const float* bias) {
    int Dm = 4 * C;
    gemm64(Abf, W, nullptr, PRE, nullptr, Nn, K, Dm);               // pre = h @ W^T
    gat_scores_kernel<<<cdiv(Nn * 4, 256), 256, 0, stream>>>(PRE, a_s, a_d, as_, ad_, C, Nn);
    fillz(mmax, Nn * 4);
    fillz(denom, Nn * 4);
    fillz(ACC, Nn * Dm);
    edge_max_kernel<<<cdiv(Et * 4, 256), 256, 0, stream>>>(ei, as_, ad_, mmax, Et, Eo);
    edge_accum_kernel<<<cdiv(Et * 4, 256), 256, 0, stream>>>(ei, as_, ad_, mmax, PRE,
                                                             denom, ACC, Et, Eo, C);
    gat_finalize_kernel<<<cdiv(Nn * Dm, 256), 256, 0, stream>>>(ACC, denom, bias,
                                                                Hf, actbf, Nn, Dm, C);
  };
  gat_layer(actbf, g1W, 32, 32, F(I_G1AS), F(I_G1AD), F(I_G1B));   // -> [N,128]
  gat_layer(actbf, g2W, 128, 16, F(I_G2AS), F(I_G2AD), F(I_G2B));  // -> [N,64]

  // ---- 4) transformer blocks ----
  auto attn_block = [&](const __bf16* inbf, const __bf16* wv, const float* bv,
                        const __bf16* wo, const float* bo,
                        const float* lg, const float* lb) {
    gemm64(inbf, wv, bv, nullptr, tmpbf, Nn, 64, 64);   // v = in @ wv^T + bv
    gemm64(tmpbf, wo, bo, Yf, nullptr, Nn, 64, 64);     // y = v  @ wo^T + bo
    ln_residual_kernel<<<cdiv(Nn, 128), 128, 0, stream>>>(Hf, Yf, lg, lb, Hf, actbf, Nn);
  };
  auto ff_block = [&](int l, const float* b1, const float* b2,
                      const float* lg, const float* lb) {
    ff_fused_kernel<<<cdiv(Nn, 128), 256, 0, stream>>>(actbf, w1W[l], b1, w2W[l], b2, Yf, Nn);
    ln_residual_kernel<<<cdiv(Nn, 128), 128, 0, stream>>>(Hf, Yf, lg, lb, Hf, actbf, Nn);
  };

  for (int l = 0; l < 2; ++l) {   // encoder
    int b = Lbase[l];
    attn_block(actbf, wvW[l], F(b + 1), woW[l], F(b + 3), F(b + 8), F(b + 9));
    ff_block(l, F(b + 5), F(b + 7), F(b + 10), F(b + 11));
  }
  hipMemcpyAsync(MEMf, Hf, (size_t)Nn * 64 * 4, hipMemcpyDeviceToDevice, stream);
  hipMemcpyAsync(membf, actbf, (size_t)Nn * 64 * 2, hipMemcpyDeviceToDevice, stream);

  for (int l = 2; l < 4; ++l) {   // decoder
    int b = Lbase[l];
    attn_block(actbf, wvW[l], F(b + 1), woW[l], F(b + 3), F(b + 8), F(b + 9));        // self, ln1
    attn_block(membf, wv2W[l], F(b + 13), wo2W[l], F(b + 15), F(b + 10), F(b + 11));  // cross, ln2
    ff_block(l, F(b + 5), F(b + 7), F(b + 16), F(b + 17));                            // ff, ln3
  }

  // ---- 5) mean-pool per graph + final FC ----
  fillz(pooled, 64 * 64);
  fillz(cntb, 64);
  pool_accum_kernel<<<cdiv(Nn, 256), 256, 0, stream>>>(Hf, batch, pooled, cntb, Nn);
  final_fc_kernel<<<cdiv(64 * 16, 256), 256, 0, stream>>>(pooled, cntb, F(I_FCW), F(I_FCB),
                                                          (float*)d_out);
}
